// CP_TransformerDecoder_Action_1563368096244
// MI455X (gfx1250) — matrix operand order repaired and checked by reading, not compiled, not run on hardware
//
#include <hip/hip_runtime.h>
#include <cstddef>
#include <cstdint>

// ---------------------------------------------------------------- types
typedef __bf16 bf16;
typedef __attribute__((ext_vector_type(8)))  __bf16 v8bf;
typedef __attribute__((ext_vector_type(16))) __bf16 v16bf;
typedef __attribute__((ext_vector_type(8)))  float  v8f;
typedef __attribute__((ext_vector_type(4)))  unsigned v4ui;
typedef __attribute__((ext_vector_type(4)))  int    v4i;
typedef __attribute__((ext_vector_type(8)))  int    v8i;

#define SHUF16(lo, hi) __builtin_shufflevector((lo), (hi), 0,1,2,3,4,5,6,7,8,9,10,11,12,13,14,15)

static __device__ __forceinline__ bf16 f2bf(float x) { return (bf16)x; }

// Problem dims (compile-time constants from the reference)
#define LNUM 4
#define BB   2
#define NN   1024
#define CC   1024
#define HH   16
#define DD   64
#define RR   64
#define HID  4096
#define MTOK 2048   // B*N rows

// ---------------------------------------------------------------- TDM: 2D tile of bf16 -> LDS
//  Loads a [tile_rows=128 x 32] bf16 tile whose rows are spaced `ldw_elems` apart.
//  tensor_dim1 = rows_valid -> TDM zero-fills rows beyond the tensor (small-N GEMMs).
static __device__ __forceinline__ void tdm_load_tile(const bf16* gsrc, unsigned ldw_elems,
                                                     unsigned rows_valid, unsigned lds_off) {
    const unsigned long long ga = (unsigned long long)(uintptr_t)gsrc;   // byte address
    v4ui g0;
    g0[0] = 1u;                                               // count=1, user mode, no gather
    g0[1] = lds_off;                                          // lds_addr (bytes)
    g0[2] = (unsigned)(ga & 0xFFFFFFFFu);                     // global_addr[31:0]
    g0[3] = (unsigned)((ga >> 32) & 0x01FFFFFFu) | (2u << 30);// global_addr[56:32] | type=2

    const unsigned long long s0 = (unsigned long long)ldw_elems;  // stride in data_size units
    v8i g1;
    g1[0] = (int)(1u << 16);                                  // workgroup_mask=0, data_size=1 (2B)
    g1[1] = (int)((32u & 0xFFFFu) << 16);                     // tensor_dim0 = 32 (lo16)
    g1[2] = (int)((rows_valid & 0xFFFFu) << 16);              // dim0 hi16=0 | tensor_dim1 lo16
    g1[3] = (int)((rows_valid >> 16) | (32u << 16));          // dim1 hi16 | tile_dim0 = 32
    g1[4] = (int)128;                                         // tile_dim1 = 128, tile_dim2 = 0
    g1[5] = (int)(unsigned)(s0 & 0xFFFFFFFFu);                // tensor_dim0_stride[31:0]
    g1[6] = (int)(unsigned)((s0 >> 32) & 0xFFFFu);            // stride0[47:32] | stride1 lo = 0
    g1[7] = 0;                                                // stride1 hi = 0

    const v4i z4 = {0, 0, 0, 0};
#if defined(__clang_major__) && (__clang_major__ >= 23)
    const v8i z8 = {0, 0, 0, 0, 0, 0, 0, 0};
    __builtin_amdgcn_tensor_load_to_lds(g0, g1, z4, z4, z8, 0);
#else
    __builtin_amdgcn_tensor_load_to_lds(g0, g1, z4, z4, 0);
#endif
}

// ---------------------------------------------------------------- convert fp32 -> bf16
__global__ void f32_to_bf16_kernel(const float* __restrict__ in, bf16* __restrict__ out,
                                   unsigned long long n) {
    unsigned long long i = (unsigned long long)blockIdx.x * 256ull + threadIdx.x;
    unsigned long long stride = (unsigned long long)gridDim.x * 256ull;
    for (; i < n; i += stride) out[i] = f2bf(in[i]);
}

// ---------------------------------------------------------------- residual add
__global__ void add_inplace_kernel(float* __restrict__ x, const float* __restrict__ y,
                                   unsigned long long n) {
    unsigned long long i = (unsigned long long)blockIdx.x * 256ull + threadIdx.x;
    unsigned long long stride = (unsigned long long)gridDim.x * 256ull;
    for (; i < n; i += stride) x[i] += y[i];
}

// ---------------------------------------------------------------- exact GELU, fp32 in -> bf16 out
__global__ void gelu_bf16_kernel(const float* __restrict__ in, bf16* __restrict__ out,
                                 unsigned long long n) {
    unsigned long long i = (unsigned long long)blockIdx.x * 256ull + threadIdx.x;
    unsigned long long stride = (unsigned long long)gridDim.x * 256ull;
    for (; i < n; i += stride) {
        float v = in[i];
        out[i] = f2bf(0.5f * v * (1.0f + erff(v * 0.70710678118654752f)));
    }
}

// ---------------------------------------------------------------- LayerNorm: fp32 row -> bf16 row
__global__ __launch_bounds__(256)
void ln_bf16_kernel(const float* __restrict__ X, const float* __restrict__ g,
                    const float* __restrict__ b, bf16* __restrict__ O) {
    __shared__ float red[256];
    const int row = blockIdx.x;
    const float* x = X + (size_t)row * CC;
    float s = 0.f;
    for (int i = threadIdx.x; i < CC; i += 256) s += x[i];
    red[threadIdx.x] = s; __syncthreads();
    for (int off = 128; off > 0; off >>= 1) {
        if ((int)threadIdx.x < off) red[threadIdx.x] += red[threadIdx.x + off];
        __syncthreads();
    }
    const float mean = red[0] * (1.0f / CC);
    __syncthreads();
    float vs = 0.f;
    for (int i = threadIdx.x; i < CC; i += 256) { float d = x[i] - mean; vs += d * d; }
    red[threadIdx.x] = vs; __syncthreads();
    for (int off = 128; off > 0; off >>= 1) {
        if ((int)threadIdx.x < off) red[threadIdx.x] += red[threadIdx.x + off];
        __syncthreads();
    }
    const float rstd = rsqrtf(red[0] * (1.0f / CC) + 1e-5f);
    bf16* o = O + (size_t)row * CC;
    for (int i = threadIdx.x; i < CC; i += 256)
        o[i] = f2bf((x[i] - mean) * rstd * g[i] + b[i]);
}

// ---------------------------------------------------------------- CP contraction:
//  Wt[(g*64+b)*64 + a]   = sum_r cp_c[a,b,r] * fac[r, g]        (g = 0..3)   -> A @ Wt^T == u @ CPc_g
//  Wd[a*256 + g*64 + b]  = sum_r cp_c[a,b,r] * fac[r, fd0+g]    (optional)   -> fc2_cp as [R, 4R]
__global__ __launch_bounds__(256)
void cp_build_kernel(const float* __restrict__ cp_c, const float* __restrict__ fac, int nf,
                     bf16* __restrict__ Wt, bf16* __restrict__ Wd, int fd0) {
    const int idx = blockIdx.x * 256 + threadIdx.x;   // 64 blocks -> 16384 = 4*64*64
    const int a = idx & 63;
    const int b = (idx >> 6) & 63;
    const int g = (idx >> 12) & 3;
    const float* c = cp_c + ((size_t)a * 64 + b) * 64;
    float st = 0.f, sd = 0.f;
    for (int r = 0; r < 64; ++r) {
        const float cv = c[r];
        st += cv * fac[r * nf + g];
        if (Wd) sd += cv * fac[r * nf + fd0 + g];
    }
    Wt[((size_t)g * 64 + b) * 64 + a] = f2bf(st);
    if (Wd) Wd[(size_t)a * 256 + g * 64 + b] = f2bf(sd);
}

// ---------------------------------------------------------------- WMMA bf16 GEMM (TDM-staged B):
//  D[M,N](fp32, ldd) (+)= A[M,K](bf16, lda) @ W[N,K]^T(bf16, ldw) (+ bias[N])
//  Block 256 threads = 8 waves (2M x 4N), block tile 64x128, wave tile 32x32.
//  W tiles [128 x 32] are DMA'd into LDS by the Tensor Data Mover (double-buffered,
//  TENSORcnt-synchronized); A fragments stream directly from global memory.
//  K multiple of 32 (>= 64); M multiple of 64; N guarded (TDM zero-fills OOB rows).
__global__ __launch_bounds__(256)
void gemm_bf16_kernel(const bf16* __restrict__ A, int lda,
                      const bf16* __restrict__ W, int ldw,
                      float* __restrict__ D, int ldd,
                      const float* __restrict__ bias,
                      int M, int N, int K, int accum) {
    __shared__ bf16 Bs[2][128 * 32];   // 2 x 8KB double buffer

    const int lane = threadIdx.x & 31;
    const int wave = threadIdx.x >> 5;
    const int wm = wave >> 2;          // 0..1
    const int wn = wave & 3;           // 0..3
    const int m0 = blockIdx.y * 64 + wm * 32;
    const int nblk = blockIdx.x * 128;
    const int n0 = nblk + wn * 32;
    const int l15 = lane & 15;
    const int lh  = lane >> 4;

    v8f acc00 = {}, acc01 = {}, acc10 = {}, acc11 = {};

    // A fragment (16x32): elems 0..7 at k = kg*8, elems 8..15 at k = 16 + kg*8  (kg = lane>>4)
    const bf16* pa0 = A + (size_t)(m0 + l15)      * lda + lh * 8;
    const bf16* pa1 = A + (size_t)(m0 + 16 + l15) * lda + lh * 8;

    const unsigned rows  = (unsigned)(N - nblk);            // valid W rows in this block's tile
    const unsigned lds0  = (unsigned)(uintptr_t)(&Bs[0][0]);
    const unsigned lds1  = (unsigned)(uintptr_t)(&Bs[1][0]);
    const bf16* wtile = W + (size_t)nblk * ldw;

    const int nsteps = K >> 5;
    if (wave == 0) tdm_load_tile(wtile, (unsigned)ldw, rows, lds0);

    for (int i = 0; i < nsteps; ++i) {
        const int kk = i << 5;
        if (wave == 0) __builtin_amdgcn_s_wait_tensorcnt(0);   // tile i resident in LDS
        __syncthreads();
        if (wave == 0 && (i + 1) < nsteps)                     // kick tile i+1 into other buffer
            tdm_load_tile(wtile + kk + 32, (unsigned)ldw, rows, (i & 1) ? lds0 : lds1);

        v16bf a0 = SHUF16(*(const v8bf*)(pa0 + kk), *(const v8bf*)(pa0 + kk + 16));
        v16bf a1 = SHUF16(*(const v8bf*)(pa1 + kk), *(const v8bf*)(pa1 + kk + 16));

        // B fragment (32x16): elems 0..7 at k = kg*8, elems 8..15 at k = 16 + kg*8 of row n (kg = lane>>4)
        const bf16* bs  = &Bs[i & 1][0];
        const bf16* pb0 = bs + ((wn * 32 + l15)      * 32 + lh * 8);
        const bf16* pb1 = bs + ((wn * 32 + 16 + l15) * 32 + lh * 8);
        v16bf b0 = SHUF16(*(const v8bf*)pb0, *(const v8bf*)(pb0 + 16));
        v16bf b1 = SHUF16(*(const v8bf*)pb1, *(const v8bf*)(pb1 + 16));

        if (kk + 32 < K) {  // stream next A lines (global_prefetch_b8)
            __builtin_prefetch(pa0 + kk + 32, 0, 3);
            __builtin_prefetch(pa1 + kk + 32, 0, 3);
        }

        acc00 = __builtin_amdgcn_wmma_f32_16x16x32_bf16(false, a0, false, b0, (short)0, acc00, false, false);
        acc01 = __builtin_amdgcn_wmma_f32_16x16x32_bf16(false, a0, false, b1, (short)0, acc01, false, false);
        acc10 = __builtin_amdgcn_wmma_f32_16x16x32_bf16(false, a1, false, b0, (short)0, acc10, false, false);
        acc11 = __builtin_amdgcn_wmma_f32_16x16x32_bf16(false, a1, false, b1, (short)0, acc11, false, false);
    }

    // C/D layout: VGPR r, lane L -> row = r + 8*(L>>4), col = L&15
    #define STORE_TILE(ACC, MT, NT)                                          \
        do {                                                                  \
            const int n = (NT) + l15;                                         \
            if (n < N) {                                                      \
                const float bz = bias ? bias[n] : 0.f;                        \
                _Pragma("unroll")                                             \
                for (int r = 0; r < 8; ++r) {                                 \
                    const int m = (MT) + 8 * lh + r;                          \
                    float v = (ACC)[r] + bz;                                  \
                    if (accum) v += D[(size_t)m * ldd + n];                   \
                    D[(size_t)m * ldd + n] = v;                               \
                }                                                             \
            }                                                                 \
        } while (0)

    STORE_TILE(acc00, m0,      n0);
    STORE_TILE(acc01, m0,      n0 + 16);
    STORE_TILE(acc10, m0 + 16, n0);
    STORE_TILE(acc11, m0 + 16, n0 + 16);
    #undef STORE_TILE
}

// ---------------------------------------------------------------- flash attention (causal)
//  qkv: fp32 [B*N, 3*C] packed as [b,n][t*1024 + h*64 + d]
//  out: fp32 [B*N, C]   at [b,n][h*64 + d]
//  grid (N/64, H, B); block 128 = 4 waves; wave handles 16 query rows.
__global__ __launch_bounds__(128)
void attention_kernel(const float* __restrict__ qkv, float* __restrict__ out) {
    __shared__ bf16 Ks[64][64];       // [key][d]
    __shared__ bf16 Vt[64][64];       // [d][key]  (transposed)
    __shared__ bf16 Ps[4][16][64];    // per-wave P tile [m_local][key]

    const int lane = threadIdx.x & 31;
    const int wave = threadIdx.x >> 5;     // 0..3
    const int l15 = lane & 15;
    const int lh  = lane >> 4;
    const int qt = blockIdx.x;
    const int h  = blockIdx.y;
    const int bb = blockIdx.z;
    const int qbase = qt * 64;
    const size_t tokbase = (size_t)bb * NN;
    const float scale = 0.125f;            // D^-0.5, folded into Q

    // Q A-fragments for this wave's 16 rows, kept in registers across the key loop
    const int mrow = qbase + wave * 16 + l15;
    const float* qptr = qkv + (tokbase + mrow) * (3 * CC) + h * DD;
    v16bf qa[2];
    #pragma unroll
    for (int kf = 0; kf < 2; ++kf) {
        const float* pb = qptr + kf * 32 + lh * 8;
        v8bf lo8, hi8;
        #pragma unroll
        for (int i = 0; i < 8; ++i) lo8[i] = f2bf(pb[i] * scale);
        #pragma unroll
        for (int i = 0; i < 8; ++i) hi8[i] = f2bf(pb[16 + i] * scale);
        qa[kf] = SHUF16(lo8, hi8);
    }

    float mrun[8], lrun[8];
    v8f oacc[4];
    #pragma unroll
    for (int r = 0; r < 8; ++r) { mrun[r] = -3.0e38f; lrun[r] = 0.f; }
    #pragma unroll
    for (int nt = 0; nt < 4; ++nt) oacc[nt] = (v8f){};

    for (int kt = 0; kt <= qt; ++kt) {
        __syncthreads();
        // cooperative K / V^T tile load (fp32 -> bf16)
        for (int e = threadIdx.x; e < 64 * 64; e += 128) {
            const int row = e >> 6, d = e & 63;
            const size_t tok = tokbase + kt * 64 + row;
            Ks[row][d] = f2bf(qkv[tok * (3 * CC) + CC     + h * DD + d]);
            Vt[d][row] = f2bf(qkv[tok * (3 * CC) + 2 * CC + h * DD + d]);
        }
        __syncthreads();

        // S = Q @ K^T   (16 rows x 64 keys per wave)
        v8f sacc[4];
        #pragma unroll
        for (int nt = 0; nt < 4; ++nt) {
            sacc[nt] = (v8f){};
            #pragma unroll
            for (int kf = 0; kf < 2; ++kf) {
                const bf16* bp = &Ks[nt * 16 + l15][kf * 32 + lh * 8];
                v16bf bfrag = SHUF16(*(const v8bf*)bp, *(const v8bf*)(bp + 16));
                sacc[nt] = __builtin_amdgcn_wmma_f32_16x16x32_bf16(
                    false, qa[kf], false, bfrag, (short)0, sacc[nt], false, false);
            }
        }

        // causal mask on the diagonal tile
        if (kt == qt) {
            #pragma unroll
            for (int nt = 0; nt < 4; ++nt) {
                const int key = kt * 64 + nt * 16 + l15;
                #pragma unroll
                for (int r = 0; r < 8; ++r) {
                    const int m = qbase + wave * 16 + 8 * lh + r;
                    if (key > m) sacc[nt][r] = -1e30f;
                }
            }
        }

        // online softmax (rows live in lanes sharing lh; reduce across 16-lane group)
        #pragma unroll
        for (int r = 0; r < 8; ++r) {
            float mx = fmaxf(fmaxf(sacc[0][r], sacc[1][r]), fmaxf(sacc[2][r], sacc[3][r]));
            #pragma unroll
            for (int off = 1; off < 16; off <<= 1) mx = fmaxf(mx, __shfl_xor(mx, off, 32));
            const float mnew = fmaxf(mrun[r], mx);
            const float corr = __expf(mrun[r] - mnew);
            float psum = 0.f;
            #pragma unroll
            for (int nt = 0; nt < 4; ++nt) {
                const float p = __expf(sacc[nt][r] - mnew);
                sacc[nt][r] = p;
                psum += p;
            }
            #pragma unroll
            for (int off = 1; off < 16; off <<= 1) psum += __shfl_xor(psum, off, 32);
            lrun[r] = lrun[r] * corr + psum;
            mrun[r] = mnew;
            #pragma unroll
            for (int nt = 0; nt < 4; ++nt) oacc[nt][r] *= corr;
        }

        // re-fragment P via LDS (C/D layout -> A layout); per-wave region, DS ops in-order
        #pragma unroll
        for (int nt = 0; nt < 4; ++nt)
            #pragma unroll
            for (int r = 0; r < 8; ++r)
                Ps[wave][8 * lh + r][nt * 16 + l15] = f2bf(sacc[nt][r]);

        v16bf pa[2];
        #pragma unroll
        for (int kf = 0; kf < 2; ++kf) {
            const bf16* pp = &Ps[wave][l15][kf * 32 + lh * 8];
            pa[kf] = SHUF16(*(const v8bf*)pp, *(const v8bf*)(pp + 16));
        }

        // O += P @ V
        #pragma unroll
        for (int nt = 0; nt < 4; ++nt) {
            #pragma unroll
            for (int kf = 0; kf < 2; ++kf) {
                const bf16* vp = &Vt[nt * 16 + l15][kf * 32 + lh * 8];
                v16bf vfrag = SHUF16(*(const v8bf*)vp, *(const v8bf*)(vp + 16));
                oacc[nt] = __builtin_amdgcn_wmma_f32_16x16x32_bf16(
                    false, pa[kf], false, vfrag, (short)0, oacc[nt], false, false);
            }
        }
    }

    // normalize and store fp32
    float* op = out + (tokbase + qbase + wave * 16) * CC + h * DD;
    #pragma unroll
    for (int r = 0; r < 8; ++r) {
        const float inv = 1.0f / lrun[r];
        const int ml = 8 * lh + r;
        #pragma unroll
        for (int nt = 0; nt < 4; ++nt)
            op[(size_t)ml * CC + nt * 16 + l15] = oacc[nt][r] * inv;
    }
}

// ================================================================ host
extern "C" void kernel_launch(void* const* d_in, const int* in_sizes, int n_in,
                              void* d_out, int out_size, void* d_ws, size_t ws_size,
                              hipStream_t stream) {
    (void)in_sizes; (void)n_in; (void)out_size; (void)ws_size;

    const float* x_in   = (const float*)d_in[0];
    // d_in[1] = mask : causal structure is known statically; not read.
    const float* ln1_g  = (const float*)d_in[2];
    const float* ln1_b  = (const float*)d_in[3];
    const float* qkv_w  = (const float*)d_in[4];
    const float* proj_w = (const float*)d_in[5];
    const float* proj_b = (const float*)d_in[6];
    const float* cp_att = (const float*)d_in[7];
    const float* ln2_g  = (const float*)d_in[8];
    const float* ln2_b  = (const float*)d_in[9];
    const float* fc1_w  = (const float*)d_in[10];
    const float* fc1_b  = (const float*)d_in[11];
    const float* fc2_w  = (const float*)d_in[12];
    const float* fc2_b  = (const float*)d_in[13];
    const float* mlp_cp = (const float*)d_in[14];
    const float* u_w    = (const float*)d_in[15];
    const float* v_w    = (const float*)d_in[16];
    const float* cp_c   = (const float*)d_in[17];

    float* xbuf = (float*)d_out;   // residual stream lives in d_out

    // -------- workspace layout (256B aligned bump allocator)
    size_t off = 0;
    auto take = [&](size_t bytes) -> void* {
        void* p = (char*)d_ws + off;
        off += (bytes + 255) & ~(size_t)255;
        return p;
    };
    bf16* wqkv  = (bf16*)take((size_t)LNUM * 3 * CC * CC * 2);
    bf16* wproj = (bf16*)take((size_t)LNUM * CC * CC * 2);
    bf16* wfc1  = (bf16*)take((size_t)LNUM * HID * CC * 2);
    bf16* wfc2  = (bf16*)take((size_t)LNUM * CC * HID * 2);
    bf16* wu    = (bf16*)take((size_t)RR * CC * 2);
    bf16* wv    = (bf16*)take((size_t)CC * RR * 2);
    bf16* wsatt = (bf16*)take((size_t)256 * 64 * 2);   // CP_att slices 0..3 (transposed)
    bf16* wsmlp = (bf16*)take((size_t)256 * 64 * 2);   // CP_mlp fc1 slices 0..3 (transposed)
    bf16* wfc2cp= (bf16*)take((size_t)64 * 256 * 2);   // fc2_cp [R,4R]
    bf16* h_bf  = (bf16*)take((size_t)MTOK * CC * 2);
    float* qkvf = (float*)take((size_t)MTOK * 3 * CC * 4);
    float* attf = (float*)take((size_t)MTOK * CC * 4);
    bf16* att_bf= (bf16*)take((size_t)MTOK * CC * 2);
    float* prf  = (float*)take((size_t)MTOK * CC * 4);
    float* hhf  = (float*)take((size_t)MTOK * HID * 4);
    bf16* a_bf  = (bf16*)take((size_t)MTOK * HID * 2);
    float* of_  = (float*)take((size_t)MTOK * CC * 4);
    float* uf   = (float*)take((size_t)MTOK * RR * 4);
    bf16* u_bf  = (bf16*)take((size_t)MTOK * RR * 2);
    float* sf   = (float*)take((size_t)MTOK * 192 * 4);
    bf16* s_bf  = (bf16*)take((size_t)MTOK * 192 * 2);
    float* s3f  = (float*)take((size_t)MTOK * RR * 4);
    bf16* s3_bf = (bf16*)take((size_t)MTOK * RR * 2);
    float* smf  = (float*)take((size_t)MTOK * 256 * 4);
    bf16* sm_bf = (bf16*)take((size_t)MTOK * 256 * 2);
    float* u4f  = (float*)take((size_t)MTOK * 256 * 4);
    bf16* u4_bf = (bf16*)take((size_t)MTOK * 256 * 2);
    float* s2f  = (float*)take((size_t)MTOK * RR * 4);
    bf16* s2_bf = (bf16*)take((size_t)MTOK * RR * 2);

    auto conv = [&](const float* in, bf16* outp, size_t n) {
        unsigned blocks = (unsigned)(((n + 255) / 256) < 4096 ? ((n + 255) / 256) : 4096);
        f32_to_bf16_kernel<<<blocks, 256, 0, stream>>>(in, outp, (unsigned long long)n);
    };
    auto gemm = [&](const bf16* A, int lda, const bf16* W, int ldw,
                    float* Dd, int ldd, const float* bias, int Nn, int K, int acc) {
        dim3 g((Nn + 127) / 128, MTOK / 64);
        gemm_bf16_kernel<<<g, 256, 0, stream>>>(A, lda, W, ldw, Dd, ldd, bias, MTOK, Nn, K, acc);
    };
    auto add = [&](float* x, const float* y, size_t n) {
        unsigned blocks = (unsigned)(((n + 255) / 256) < 4096 ? ((n + 255) / 256) : 4096);
        add_inplace_kernel<<<blocks, 256, 0, stream>>>(x, y, (unsigned long long)n);
    };

    // residual stream = input x
    hipMemcpyAsync(xbuf, x_in, (size_t)MTOK * CC * sizeof(float), hipMemcpyDeviceToDevice, stream);

    // weights -> bf16 (stateless: redone every call)
    conv(qkv_w,  wqkv,  (size_t)LNUM * 3 * CC * CC);
    conv(proj_w, wproj, (size_t)LNUM * CC * CC);
    conv(fc1_w,  wfc1,  (size_t)LNUM * HID * CC);
    conv(fc2_w,  wfc2,  (size_t)LNUM * CC * HID);
    conv(u_w,    wu,    (size_t)RR * CC);
    conv(v_w,    wv,    (size_t)CC * RR);

    for (int l = 0; l < LNUM; ++l) {
        const bf16* wqkv_l = wqkv + (size_t)l * 3 * CC * CC;
        const bf16* wproj_l = wproj + (size_t)l * CC * CC;
        const bf16* wfc1_l = wfc1 + (size_t)l * HID * CC;
        const bf16* wfc2_l = wfc2 + (size_t)l * CC * HID;

        // CP-contracted adapter weights for this layer
        cp_build_kernel<<<64, 256, 0, stream>>>(cp_c, cp_att + (size_t)l * RR * 4, 4, wsatt, nullptr, 0);
        cp_build_kernel<<<64, 256, 0, stream>>>(cp_c, mlp_cp + (size_t)l * RR * 8, 8, wsmlp, wfc2cp, 4);

        // ===== attention block =====
        ln_bf16_kernel<<<MTOK, 256, 0, stream>>>(xbuf, ln1_g + l * CC, ln1_b + l * CC, h_bf);
        gemm(h_bf, CC, wqkv_l, CC, qkvf, 3 * CC, nullptr, 3 * CC, CC, 0);      // qkv = h @ qkv_w^T
        gemm(h_bf, CC, wu, CC, uf, RR, nullptr, RR, CC, 0);                    // u = h @ u_w^T
        conv(uf, u_bf, (size_t)MTOK * RR);
        gemm(u_bf, RR, wsatt, RR, sf, 192, nullptr, 192, RR, 0);               // s_i = u @ CPc_i (i=0..2)
        conv(sf, s_bf, (size_t)MTOK * 192);
        for (int i = 0; i < 3; ++i)                                            // qkv += s_i @ v_w^T
            gemm(s_bf + i * RR, 192, wv, RR, qkvf + i * CC, 3 * CC, nullptr, CC, RR, 1);

        attention_kernel<<<dim3(NN / 64, HH, BB), 128, 0, stream>>>(qkvf, attf);
        conv(attf, att_bf, (size_t)MTOK * CC);

        gemm(att_bf, CC, wproj_l, CC, prf, CC, proj_b + l * CC, CC, CC, 0);    // pr = out @ proj_w^T + b
        gemm(att_bf, CC, wu, CC, uf, RR, nullptr, RR, CC, 0);                  // u2 = out @ u_w^T
        conv(uf, u_bf, (size_t)MTOK * RR);
        gemm(u_bf, RR, wsatt + 192 * 64, RR, s3f, RR, nullptr, RR, RR, 0);     // s3 = u2 @ CPc_3
        conv(s3f, s3_bf, (size_t)MTOK * RR);
        gemm(s3_bf, RR, wv, RR, prf, CC, nullptr, CC, RR, 1);                  // pr += s3 @ v_w^T
        add(xbuf, prf, (size_t)MTOK * CC);                                     // x += pr

        // ===== FFN block =====
        ln_bf16_kernel<<<MTOK, 256, 0, stream>>>(xbuf, ln2_g + l * CC, ln2_b + l * CC, h_bf);
        gemm(h_bf, CC, wfc1_l, CC, hhf, HID, fc1_b + l * HID, HID, CC, 0);     // hh = h @ fc1_w^T + b
        gemm(h_bf, CC, wu, CC, uf, RR, nullptr, RR, CC, 0);                    // u = h @ u_w^T
        conv(uf, u_bf, (size_t)MTOK * RR);
        gemm(u_bf, RR, wsmlp, RR, smf, 256, nullptr, 256, RR, 0);              // sm = u @ fc1_cp (4 groups)
        conv(smf, sm_bf, (size_t)MTOK * 256);
        for (int g = 0; g < 4; ++g)                                            // hh_g += sm_g @ v_w^T
            gemm(sm_bf + g * RR, 256, wv, RR, hhf + g * CC, HID, nullptr, CC, RR, 1);

        gelu_bf16_kernel<<<4096, 256, 0, stream>>>(hhf, a_bf, (unsigned long long)MTOK * HID);

        gemm(a_bf, HID, wfc2_l, HID, of_, CC, fc2_b + l * CC, CC, HID, 0);     // o = a @ fc2_w^T + b
        for (int g = 0; g < 4; ++g)                                            // u4_g = a_g @ u_w^T
            gemm(a_bf + g * CC, HID, wu, CC, u4f + g * RR, 256, nullptr, RR, CC, 0);
        conv(u4f, u4_bf, (size_t)MTOK * 256);
        gemm(u4_bf, 256, wfc2cp, 256, s2f, RR, nullptr, RR, 256, 0);           // s2 = u4 @ fc2_cp^T
        conv(s2f, s2_bf, (size_t)MTOK * RR);
        gemm(s2_bf, RR, wv, RR, of_, CC, nullptr, CC, RR, 1);                  // o += s2 @ v_w^T
        add(xbuf, of_, (size_t)MTOK * CC);                                     // x += o
    }
}
